// Qwen3CopiedAttention_16183436771570
// MI455X (gfx1250) — matrix operand order repaired and checked by reading, not compiled, not run on hardware
//
#include <hip/hip_runtime.h>
#include <cstdint>
#include <cstddef>

// ---------------- problem constants ----------------
constexpr int BB = 2, SS = 2048, DD = 2048, HH = 16, KV = 8, HD = 128;
constexpr float EPS    = 1e-6f;
constexpr float NEGM   = -1000000000.0f;
constexpr float SCALE  = 0.08838834764831845f;   // HD^-0.5
constexpr float LOG2E  = 1.4426950408889634f;

// ---------------- WMMA types ----------------
typedef __attribute__((ext_vector_type(16))) __bf16 v16bf;
typedef __attribute__((ext_vector_type(8)))  float  v8f;

union Frag16 {
    uint4 q[2];
    v16bf v;
    unsigned short us[16];
};

static __device__ __forceinline__ v8f zero8() {
    v8f z;
#pragma unroll
    for (int i = 0; i < 8; ++i) z[i] = 0.0f;
    return z;
}

static __device__ __forceinline__ unsigned short f2bfbits(float f) {
    uint32_t u = __float_as_uint(f);
    uint32_t r = u + 0x7FFFu + ((u >> 16) & 1u);   // round-to-nearest-even
    return (unsigned short)(r >> 16);
}

// ---------------- CDNA5 async global->LDS copy (ASYNCcnt DMA) ----------------
#if __has_builtin(__builtin_amdgcn_global_load_async_to_lds_b128) && \
    __has_builtin(__builtin_amdgcn_s_wait_asynccnt)
#define HAS_ASYNC_LDS 1
#else
#define HAS_ASYNC_LDS 0
#endif

#if HAS_ASYNC_LDS
// exact pointee type from the compiler diagnostic: 16B int vector
typedef int v4i_vs __attribute__((vector_size(4 * sizeof(int))));
typedef __attribute__((address_space(1))) v4i_vs GV4;   // global (__device__)
typedef __attribute__((address_space(3))) v4i_vs LV4;   // LDS
static __device__ __forceinline__ void async_copy16(const void* g, void* l) {
    // flat LDS pointer low 32 bits == LDS byte offset (aperture mapping)
    __builtin_amdgcn_global_load_async_to_lds_b128(
        (GV4*)(unsigned long long)g,
        (LV4*)(unsigned int)(unsigned long long)l,
        0, 0);
}
static __device__ __forceinline__ void async_wait0() {
    __builtin_amdgcn_s_wait_asynccnt(0);
}
#else
static __device__ __forceinline__ void async_copy16(const void* g, void* l) {
    *(uint4*)l = *(const uint4*)g;
}
static __device__ __forceinline__ void async_wait0() {}
#endif

// =====================================================================
// Kernel 1a: elementwise fp32 -> bf16 (n divisible by 4)
// =====================================================================
__global__ __launch_bounds__(256) void k_convert_bf16(const float* __restrict__ src,
                                                      unsigned short* __restrict__ dst,
                                                      long n4) {
    long i = (long)blockIdx.x * blockDim.x + threadIdx.x;
    if (i >= n4) return;
    const float4 x = reinterpret_cast<const float4*>(src)[i];
    union { unsigned short h[4]; uint2 u2; } p;
    p.h[0] = f2bfbits(x.x); p.h[1] = f2bfbits(x.y);
    p.h[2] = f2bfbits(x.z); p.h[3] = f2bfbits(x.w);
    reinterpret_cast<uint2*>(dst)[i] = p.u2;
}

// =====================================================================
// Kernel 1b: transpose + convert: src[R][C] f32 -> dst[C][R] bf16
// =====================================================================
__global__ __launch_bounds__(256) void k_transpose_bf16(const float* __restrict__ src,
                                                        unsigned short* __restrict__ dst,
                                                        int R, int C) {
    __shared__ float tile[32][33];
    const int tx = threadIdx.x & 31;
    const int ty = threadIdx.x >> 5;
    const int c0 = blockIdx.x * 32;
    const int r0 = blockIdx.y * 32;
#pragma unroll
    for (int k = 0; k < 4; ++k)
        tile[ty + k * 8][tx] = src[(size_t)(r0 + ty + k * 8) * C + c0 + tx];
    __syncthreads();
#pragma unroll
    for (int k = 0; k < 4; ++k)
        dst[(size_t)(c0 + ty + k * 8) * R + r0 + tx] = f2bfbits(tile[tx][ty + k * 8]);
}

// =====================================================================
// Kernel 2: bf16 GEMM  C[M][N](f32) = A[M][K] * BT[N][K]^T
// 256 threads (8 waves), block tile 128x128, K-step 32.
// Double-buffered LDS, async global->LDS DMA, ONE barrier per K-step.
// =====================================================================
__global__ __launch_bounds__(256) void k_gemm_bf16(const unsigned short* __restrict__ A,
                                                   const unsigned short* __restrict__ BT,
                                                   float* __restrict__ C,
                                                   int M, int N, int K) {
    __shared__ unsigned short As[2][128][40];   // 80B row stride (16B aligned)
    __shared__ unsigned short Bs[2][128][40];

    const int tid  = threadIdx.x;
    const int lane = tid & 31;
    const int wid  = tid >> 5;
    const int wm   = wid >> 1;      // 0..3
    const int wn   = wid & 1;       // 0..1
    const int l16  = lane & 15;
    const int half = lane >> 4;
    const int m0   = blockIdx.y * 128;
    const int n0   = blockIdx.x * 128;

    v8f acc[2][4];
#pragma unroll
    for (int mt = 0; mt < 2; ++mt)
#pragma unroll
        for (int nt = 0; nt < 4; ++nt) acc[mt][nt] = zero8();

    // stage one 128x32 K-slice of A and BT into LDS buffer `buf`
    auto stage = [&](int buf, int k0) {
#pragma unroll
        for (int i = 0; i < 2; ++i) {
            const int c  = tid + i * 256;
            const int r  = c >> 2;
            const int kc = (c & 3) << 3;
            async_copy16(A  + (size_t)(m0 + r) * K + k0 + kc, &As[buf][r][kc]);
            async_copy16(BT + (size_t)(n0 + r) * K + k0 + kc, &Bs[buf][r][kc]);
        }
    };

    stage(0, 0);   // prologue

    for (int k0 = 0; k0 < K; k0 += 32) {
        const int cur = (k0 >> 5) & 1;
        async_wait0();          // my DMA for `cur` tile (and older) complete
        __syncthreads();        // everyone's `cur` ready; prev reads of `cur^1` done
        if (k0 + 32 < K) stage(cur ^ 1, k0 + 32);   // DMA next tile during compute

        Frag16 a[2], b[4];
        const int kb = half * 8;
#pragma unroll
        for (int mt = 0; mt < 2; ++mt) {
            const int r = wm * 32 + mt * 16 + l16;
            a[mt].q[0] = *(const uint4*)(&As[cur][r][kb]);
            a[mt].q[1] = *(const uint4*)(&As[cur][r][kb + 16]);
        }
        const int kh = half * 8;
#pragma unroll
        for (int nt = 0; nt < 4; ++nt) {
            const int r = wn * 64 + nt * 16 + l16;
            b[nt].q[0] = *(const uint4*)(&Bs[cur][r][kh]);
            b[nt].q[1] = *(const uint4*)(&Bs[cur][r][kh + 16]);
        }
#pragma unroll
        for (int mt = 0; mt < 2; ++mt)
#pragma unroll
            for (int nt = 0; nt < 4; ++nt)
                acc[mt][nt] = __builtin_amdgcn_wmma_f32_16x16x32_bf16(
                    false, a[mt].v, false, b[nt].v, (short)0, acc[mt][nt], false, false);
    }

    // ---- epilogue ----
#pragma unroll
    for (int mt = 0; mt < 2; ++mt)
#pragma unroll
        for (int nt = 0; nt < 4; ++nt) {
            const int col = n0 + wn * 64 + nt * 16 + l16;
            const int rb  = m0 + wm * 32 + mt * 16 + half * 8;
#pragma unroll
            for (int r = 0; r < 8; ++r)
                C[(size_t)(rb + r) * N + col] = acc[mt][nt][r];
        }
}

// =====================================================================
// Kernel 3: RMSNorm + RoPE (q,k) / convert (v).  One wave32 per 128-row.
// =====================================================================
__global__ __launch_bounds__(128) void k_norm_rope(const float* __restrict__ qraw,
                                                   const float* __restrict__ kraw,
                                                   const float* __restrict__ vraw,
                                                   const float* __restrict__ cosb,
                                                   const float* __restrict__ sinb,
                                                   const float* __restrict__ qw,
                                                   const float* __restrict__ kw,
                                                   unsigned short* __restrict__ qbf,
                                                   unsigned short* __restrict__ kbf,
                                                   unsigned short* __restrict__ vbf) {
    const int lane = threadIdx.x & 31;
    const int wid  = threadIdx.x >> 5;
    const long idx = (long)blockIdx.x * 4 + wid;

    const long RQ = (long)BB * SS * HH;
    const long RK = (long)BB * SS * KV;

    const int d0 = lane * 4;
    union { unsigned short h[4]; uint2 u2; } pk;

    if (idx < RQ + RK) {
        const bool isQ = (idx < RQ);
        long t = isQ ? idx : idx - RQ;
        const int nh = isQ ? HH : KV;
        const int h  = (int)(t % nh); t /= nh;
        const int s  = (int)(t % SS);
        const int b  = (int)(t / SS);
        const float* src = (isQ ? qraw : kraw) + ((size_t)(b * SS + s)) * (nh * HD) + (size_t)h * HD;
        const float* w   = isQ ? qw : kw;
        unsigned short* dst = (isQ ? qbf : kbf) + (((size_t)(b * nh + h)) * SS + s) * HD;

        const float4 x = *reinterpret_cast<const float4*>(src + d0);
        float ss = x.x * x.x + x.y * x.y + x.z * x.z + x.w * x.w;
#pragma unroll
        for (int off = 16; off > 0; off >>= 1) ss += __shfl_xor(ss, off, 32);
        const float inv = __frsqrt_rn(ss * (1.0f / HD) + EPS);

        const int dp = (d0 + 64) & 127;
        const float4 xp = *reinterpret_cast<const float4*>(src + dp);
        const float4 w0 = *reinterpret_cast<const float4*>(w + d0);
        const float4 wp = *reinterpret_cast<const float4*>(w + dp);
        const float sgn = (d0 < 64) ? -1.0f : 1.0f;

        const size_t cso = ((size_t)b * SS + s) * HD + d0;
        const float4 c4 = *reinterpret_cast<const float4*>(cosb + cso);
        const float4 s4 = *reinterpret_cast<const float4*>(sinb + cso);

        const float xn[4] = { x.x * inv * w0.x,  x.y * inv * w0.y,  x.z * inv * w0.z,  x.w * inv * w0.w };
        const float xr[4] = { sgn * xp.x * inv * wp.x, sgn * xp.y * inv * wp.y,
                              sgn * xp.z * inv * wp.z, sgn * xp.w * inv * wp.w };
        const float cc[4] = { c4.x, c4.y, c4.z, c4.w };
        const float sn[4] = { s4.x, s4.y, s4.z, s4.w };
#pragma unroll
        for (int e = 0; e < 4; ++e) pk.h[e] = f2bfbits(xn[e] * cc[e] + xr[e] * sn[e]);
        *reinterpret_cast<uint2*>(dst + d0) = pk.u2;
    } else {
        long t = idx - RQ - RK;
        const int h = (int)(t % KV); t /= KV;
        const int s = (int)(t % SS);
        const int b = (int)(t / SS);
        const float* src = vraw + ((size_t)(b * SS + s)) * (KV * HD) + (size_t)h * HD;
        unsigned short* dst = vbf + (((size_t)(b * KV + h)) * SS + s) * HD;
        const float4 x = *reinterpret_cast<const float4*>(src + d0);
        pk.h[0] = f2bfbits(x.x); pk.h[1] = f2bfbits(x.y);
        pk.h[2] = f2bfbits(x.z); pk.h[3] = f2bfbits(x.w);
        *reinterpret_cast<uint2*>(dst + d0) = pk.u2;
    }
}

// =====================================================================
// Kernel 4: causal flash attention (GQA).  grid = (S/64, B*H), block 128.
// K tile staged via async DMA; V staged with an in-LDS transpose.
// =====================================================================
__global__ __launch_bounds__(128) void k_flash_attn(const unsigned short* __restrict__ qbf,
                                                    const unsigned short* __restrict__ kbf,
                                                    const unsigned short* __restrict__ vbf,
                                                    unsigned short* __restrict__ attn) {
    __shared__ unsigned short Ks[64][136];     // K rows, d-contig (272B stride)
    __shared__ unsigned short Vt[128][72];     // V transposed: [d][krow] (144B stride)
    __shared__ unsigned short Ps[4][16][72];   // per-wave P staging

    const int tid  = threadIdx.x;
    const int lane = tid & 31;
    const int wid  = tid >> 5;
    const int l16  = lane & 15;
    const int half = lane >> 4;

    const int bh = blockIdx.y;
    const int b  = bh / HH;
    const int h  = bh % HH;
    const int hk = h / (HH / KV);
    const int qt = blockIdx.x;
    const int q0 = qt * 64 + wid * 16;

    const unsigned short* qbase = qbf + ((size_t)(b * HH + h)  * SS) * HD;
    const unsigned short* kbase = kbf + ((size_t)(b * KV + hk) * SS) * HD;
    const unsigned short* vbase = vbf + ((size_t)(b * KV + hk) * SS) * HD;

    // ---- Q fragments in registers for the whole row-block ----
    Frag16 qa[4];
    {
        const unsigned short* qrow = qbase + (size_t)(q0 + l16) * HD;
#pragma unroll
        for (int ks = 0; ks < 4; ++ks) {
            const int kb = ks * 32 + half * 8;
            qa[ks].q[0] = *(const uint4*)(qrow + kb);
            qa[ks].q[1] = *(const uint4*)(qrow + kb + 16);
        }
    }

    v8f o[8];
#pragma unroll
    for (int dt = 0; dt < 8; ++dt) o[dt] = zero8();
    float mrow[8], lrow[8];
#pragma unroll
    for (int r = 0; r < 8; ++r) { mrow[r] = -3.0e38f; lrow[r] = 0.0f; }

    const int nblk = qt + 1;           // causal: skip blocks above the diagonal
    for (int blk = 0; blk < nblk; ++blk) {
        const int kb0 = blk * 64;
        __syncthreads();
        // ---- stage K via async DMA, V via register-transpose scatter ----
#pragma unroll
        for (int i = 0; i < 8; ++i) {
            const int c  = tid + i * 128;
            const int r  = c >> 4;
            const int dc = (c & 15) << 3;
            async_copy16(kbase + (size_t)(kb0 + r) * HD + dc, &Ks[r][dc]);
            union { uint4 q; unsigned short h[8]; } vv;
            vv.q = *(const uint4*)(vbase + (size_t)(kb0 + r) * HD + dc);
#pragma unroll
            for (int e = 0; e < 8; ++e) Vt[dc + e][r] = vv.h[e];
        }
        async_wait0();
        __syncthreads();

        // ---- scores: S = Q (16x128) * K^T (128x64) ----
        v8f sc[4];
#pragma unroll
        for (int nt = 0; nt < 4; ++nt) {
            sc[nt] = zero8();
#pragma unroll
            for (int ks = 0; ks < 4; ++ks) {
                Frag16 bb;
                const int r  = nt * 16 + l16;
                const int kh = ks * 32 + half * 8;
                bb.q[0] = *(const uint4*)(&Ks[r][kh]);
                bb.q[1] = *(const uint4*)(&Ks[r][kh + 16]);
                sc[nt] = __builtin_amdgcn_wmma_f32_16x16x32_bf16(
                    false, qa[ks].v, false, bb.v, (short)0, sc[nt], false, false);
            }
        }

        // ---- scale + causal mask ----
#pragma unroll
        for (int nt = 0; nt < 4; ++nt) {
            const int col = kb0 + nt * 16 + l16;
#pragma unroll
            for (int r = 0; r < 8; ++r) {
                const int row = q0 + half * 8 + r;
                float s = sc[nt][r] * SCALE;
                if (col > row) s += NEGM;
                sc[nt][r] = s;
            }
        }

        // ---- online softmax (row stats across 16 lanes of this half) ----
#pragma unroll
        for (int r = 0; r < 8; ++r) {
            float mx = fmaxf(fmaxf(sc[0][r], sc[1][r]), fmaxf(sc[2][r], sc[3][r]));
#pragma unroll
            for (int off = 8; off > 0; off >>= 1) mx = fmaxf(mx, __shfl_xor(mx, off, 32));
            const float mnew = fmaxf(mrow[r], mx);
            const float resc = exp2f((mrow[r] - mnew) * LOG2E);
            mrow[r] = mnew;
            lrow[r] *= resc;
#pragma unroll
            for (int dt = 0; dt < 8; ++dt) o[dt][r] *= resc;
            float rs = 0.0f;
#pragma unroll
            for (int nt = 0; nt < 4; ++nt) {
                const float p = exp2f((sc[nt][r] - mnew) * LOG2E);
                sc[nt][r] = p;
                rs += p;
            }
#pragma unroll
            for (int off = 8; off > 0; off >>= 1) rs += __shfl_xor(rs, off, 32);
            lrow[r] += rs;
        }

        // ---- stage P (C-layout -> row-major) for A-fragment reload ----
#pragma unroll
        for (int nt = 0; nt < 4; ++nt)
#pragma unroll
            for (int r = 0; r < 8; ++r)
                Ps[wid][half * 8 + r][nt * 16 + l16] = f2bfbits(sc[nt][r]);
        __syncthreads();

        // ---- O += P (16x64) * V (64x128) ----
#pragma unroll
        for (int dt = 0; dt < 8; ++dt) {
#pragma unroll
            for (int ks = 0; ks < 2; ++ks) {
                Frag16 pa, vb;
                const int pkb = ks * 32 + half * 8;
                pa.q[0] = *(const uint4*)(&Ps[wid][l16][pkb]);
                pa.q[1] = *(const uint4*)(&Ps[wid][l16][pkb + 16]);
                const int dcol = dt * 16 + l16;
                const int kh   = ks * 32 + half * 8;
                vb.q[0] = *(const uint4*)(&Vt[dcol][kh]);
                vb.q[1] = *(const uint4*)(&Vt[dcol][kh + 16]);
                o[dt] = __builtin_amdgcn_wmma_f32_16x16x32_bf16(
                    false, pa.v, false, vb.v, (short)0, o[dt], false, false);
            }
        }
    }

    // ---- normalize + write attn[b][row][h*HD + d] (bf16) ----
#pragma unroll
    for (int dt = 0; dt < 8; ++dt) {
        const int d = dt * 16 + l16;
#pragma unroll
        for (int r = 0; r < 8; ++r) {
            const int row = q0 + half * 8 + r;
            const float val = o[dt][r] / lrow[r];
            attn[((size_t)(b * SS + row)) * (HH * HD) + (size_t)h * HD + d] = f2bfbits(val);
        }
    }
}

// =====================================================================
// launcher
// =====================================================================
extern "C" void kernel_launch(void* const* d_in, const int* in_sizes, int n_in,
                              void* d_out, int out_size, void* d_ws, size_t ws_size,
                              hipStream_t stream) {
    (void)in_sizes; (void)n_in; (void)out_size; (void)ws_size;
    const float* X    = (const float*)d_in[0];
    const float* cosb = (const float*)d_in[1];
    const float* sinb = (const float*)d_in[2];
    // d_in[3] attention_mask: reproduced analytically (pure causal), not read
    const float* Wq   = (const float*)d_in[4];
    const float* Wk   = (const float*)d_in[5];
    const float* Wv   = (const float*)d_in[6];
    const float* Wo   = (const float*)d_in[7];
    const float* qw   = (const float*)d_in[8];
    const float* kw   = (const float*)d_in[9];
    float* out = (float*)d_out;

    const int M = BB * SS;           // 4096
    char* ws = (char*)d_ws;
    size_t off = 0;
    auto alloc = [&](size_t bytes) { char* p = ws + off; off += (bytes + 255) & ~size_t(255); return p; };

    unsigned short* Xbf  = (unsigned short*)alloc((size_t)M * DD * 2);
    unsigned short* WqT  = (unsigned short*)alloc((size_t)(HH * HD) * DD * 2);
    unsigned short* WkT  = (unsigned short*)alloc((size_t)(KV * HD) * DD * 2);
    unsigned short* WvT  = (unsigned short*)alloc((size_t)(KV * HD) * DD * 2);
    unsigned short* WoT  = (unsigned short*)alloc((size_t)DD * (HH * HD) * 2);
    float*          qraw = (float*)alloc((size_t)M * HH * HD * 4);
    float*          kraw = (float*)alloc((size_t)M * KV * HD * 4);
    float*          vraw = (float*)alloc((size_t)M * KV * HD * 4);
    unsigned short* qbf  = (unsigned short*)alloc((size_t)M * HH * HD * 2);
    unsigned short* kbf  = (unsigned short*)alloc((size_t)M * KV * HD * 2);
    unsigned short* vbf  = (unsigned short*)alloc((size_t)M * KV * HD * 2);
    unsigned short* abf  = (unsigned short*)alloc((size_t)M * HH * HD * 2);

    // 1) conversions
    {
        const long n4 = (long)M * DD / 4;
        k_convert_bf16<<<dim3((unsigned)((n4 + 255) / 256)), dim3(256), 0, stream>>>(X, Xbf, n4);
        k_transpose_bf16<<<dim3((HH * HD) / 32, DD / 32), dim3(256), 0, stream>>>(Wq, WqT, DD, HH * HD);
        k_transpose_bf16<<<dim3((KV * HD) / 32, DD / 32), dim3(256), 0, stream>>>(Wk, WkT, DD, KV * HD);
        k_transpose_bf16<<<dim3((KV * HD) / 32, DD / 32), dim3(256), 0, stream>>>(Wv, WvT, DD, KV * HD);
        k_transpose_bf16<<<dim3(DD / 32, (HH * HD) / 32), dim3(256), 0, stream>>>(Wo, WoT, HH * HD, DD);
    }
    // 2) QKV projections
    k_gemm_bf16<<<dim3((HH * HD) / 128, M / 128), dim3(256), 0, stream>>>(Xbf, WqT, qraw, M, HH * HD, DD);
    k_gemm_bf16<<<dim3((KV * HD) / 128, M / 128), dim3(256), 0, stream>>>(Xbf, WkT, kraw, M, KV * HD, DD);
    k_gemm_bf16<<<dim3((KV * HD) / 128, M / 128), dim3(256), 0, stream>>>(Xbf, WvT, vraw, M, KV * HD, DD);
    // 3) RMSNorm + RoPE + layout
    {
        const long rows = (long)BB * SS * (HH + KV + KV);
        k_norm_rope<<<dim3((unsigned)(rows / 4)), dim3(128), 0, stream>>>(
            qraw, kraw, vraw, cosb, sinb, qw, kw, qbf, kbf, vbf);
    }
    // 4) flash attention
    k_flash_attn<<<dim3(SS / 64, BB * HH), dim3(128), 0, stream>>>(qbf, kbf, vbf, abf);
    // 5) output projection -> fp32 result
    k_gemm_bf16<<<dim3(DD / 128, M / 128), dim3(256), 0, stream>>>(abf, WoT, out, M, DD, DD);
}
